// TransformerEncoderLayer_13477607375359
// MI455X (gfx1250) — matrix operand order repaired and checked by reading, not compiled, not run on hardware
//
#include <hip/hip_runtime.h>
#include <hip/hip_bf16.h>
#include <math.h>

typedef __attribute__((ext_vector_type(16))) __bf16 v16bf;
typedef __attribute__((ext_vector_type(8)))  __bf16 v8bf;
typedef __attribute__((ext_vector_type(8)))  float  v8f;

#define WMMA_BF16(a, b, c) \
  __builtin_amdgcn_wmma_f32_16x16x32_bf16(false, (a), false, (b), (short)0, (c), false, false)

#define NH    8
#define HDIM  64
#define PMEM  16
#define CCH   512
#define HIDD  2048
#define NB    8
#define LSEQ  1024
#define LKV   1040   /* LSEQ + PMEM */

static __device__ inline v8f v8f_zero() {
  v8f z;
  for (int i = 0; i < 8; i++) z[i] = 0.0f;
  return z;
}
static __device__ inline v8bf v8bf_zero() {
  v8bf z;
  for (int i = 0; i < 8; i++) z[i] = (__bf16)0.0f;
  return z;
}
static __device__ inline v16bf combine16(v8bf lo, v8bf hi) {
  return __builtin_shufflevector(lo, hi, 0, 1, 2, 3, 4, 5, 6, 7,
                                         8, 9, 10, 11, 12, 13, 14, 15);
}

/* ================================================================== */
/* Prep kernels: swizzle all B operands into per-lane WMMA tile order. */
/* A 32x16 B-tile is stored as 32 lanes x 16 contiguous bf16 (1KB):    */
/*   element j of lane L = B[32*s + 16*(L>=16) + j][16*ct + (L&15)]    */
/* ================================================================== */

__global__ __launch_bounds__(256) void prep_small_kernel(
    const float* __restrict__ Wq, const float* __restrict__ Wk,
    const float* __restrict__ Wv, const float* __restrict__ Wo,
    const float* __restrict__ p_keys, const float* __restrict__ p_values,
    __bf16* __restrict__ WqS, __bf16* __restrict__ WkS, __bf16* __restrict__ WvS,
    __bf16* __restrict__ WoS, __bf16* __restrict__ k_sw, __bf16* __restrict__ v_sw) {
  int i = blockIdx.x * 256 + threadIdx.x;

  if (i < 64 * 64) {           /* Wq/Wk/Wv: K=64 (s=0..1), O=64 (ct=0..3) */
    int j = i & 15, lane = (i >> 4) & 31, tile = i >> 9;
    int ct = tile & 3, s = tile >> 2;
    int k = s * 32 + (lane >> 4) * 16 + j;
    int o = ct * 16 + (lane & 15);
    WqS[i] = (__bf16)Wq[o * 64 + k];
    WkS[i] = (__bf16)Wk[o * 64 + k];
    WvS[i] = (__bf16)Wv[o * 64 + k];
  }
  if (i < CCH * CCH) {         /* Wo^T: K=512 (s=0..15), O=512 (ct=0..31) */
    int j = i & 15, lane = (i >> 4) & 31, tile = i >> 9;
    int ct = tile & 31, s = tile >> 5;
    int k = s * 32 + (lane >> 4) * 16 + j;
    int o = ct * 16 + (lane & 15);
    WoS[i] = (__bf16)Wo[o * CCH + k];
  }
  if (i < NB * NH * PMEM * HDIM) {   /* persistent keys into k_sw (ct=64) */
    int d = i & 63, p = (i >> 6) & 15, h = (i >> 10) & 7, n = i >> 13;
    int pos = LSEQ + p;
    int sB = d >> 5;
    int laneB = ((d >> 4) & 1) * 16 + (pos & 15);
    int jB = d & 15;
    size_t addr = ((size_t)(n * NH + h) * 130 + sB * 65 + 64) * 512 + laneB * 16 + jB;
    k_sw[addr] = (__bf16)p_keys[p * HDIM + d];
  }
  if (i < 2 * NB * NH * PMEM * HDIM) {  /* persistent values + zero tail */
    int z = i >> 16;             /* z=0: pos 1024..1039, z=1: zero 1040..1055 */
    int ii = i & 65535;
    int d = ii & 63, p = (ii >> 6) & 15, h = (ii >> 10) & 7, n = ii >> 13;
    int pos = LSEQ + z * 16 + p;
    int dt = d >> 4;
    int laneB = ((pos >> 4) & 1) * 16 + (d & 15);
    int jB = pos & 15;
    size_t addr = ((size_t)(n * NH + h) * 132 + 32 * 4 + dt) * 512 + laneB * 16 + jB;
    v_sw[addr] = z ? (__bf16)0.0f : (__bf16)p_values[p * HDIM + d];
  }
}

/* conv1_w (4096,512,3) -> B1s swizzled: K=1536 (s=0..47), O=4096 (ct=0..255) */
__global__ __launch_bounds__(256) void prep_b1_kernel(
    const float* __restrict__ conv1_w, __bf16* __restrict__ B1s) {
  size_t i = (size_t)blockIdx.x * 256 + threadIdx.x;   /* 6291456 total */
  int j = (int)(i & 15), lane = (int)((i >> 4) & 31);
  int tile = (int)(i >> 9);
  int ct = tile & 255, s = tile >> 8;
  int k = s * 32 + (lane >> 4) * 16 + j;
  int o = ct * 16 + (lane & 15);
  int t = k / 512, c = k & 511;
  B1s[i] = (__bf16)conv1_w[((size_t)o * 512 + c) * 3 + t];
}

/* conv2_w (512,2048,3) -> B2s swizzled: K=6144 (s=0..191), O=512 (ct=0..31) */
__global__ __launch_bounds__(256) void prep_b2_kernel(
    const float* __restrict__ conv2_w, __bf16* __restrict__ B2s) {
  size_t i = (size_t)blockIdx.x * 256 + threadIdx.x;   /* 3145728 total */
  int j = (int)(i & 15), lane = (int)((i >> 4) & 31);
  int tile = (int)(i >> 9);
  int ct = tile & 31, s = tile >> 5;
  int k = s * 32 + (lane >> 4) * 16 + j;
  int o = ct * 16 + (lane & 15);
  int t = k >> 11, c = k & 2047;
  B2s[i] = (__bf16)conv2_w[((size_t)o * HIDD + c) * 3 + t];
}

/* ================================================================== */
/* K1: per-head-dim QKV projection.  Block = 16-row tile, wave = head. */
/* Writes q row-major; K and V in attention B-tile swizzled order.     */
/* ================================================================== */

static __device__ inline void wmma_proj(const v16bf a0, const v16bf a1,
                                        const __bf16* __restrict__ Wsw,
                                        int lane, v8f acc[4]) {
  for (int t = 0; t < 4; t++) {
    acc[t] = v8f_zero();
    v16bf b0 = *(const v16bf*)(Wsw + (size_t)(0 * 4 + t) * 512 + lane * 16);
    acc[t] = WMMA_BF16(a0, b0, acc[t]);
    v16bf b1 = *(const v16bf*)(Wsw + (size_t)(1 * 4 + t) * 512 + lane * 16);
    acc[t] = WMMA_BF16(a1, b1, acc[t]);
  }
}

__global__ __launch_bounds__(256) void qkv_kernel(
    const float* __restrict__ x, const __bf16* __restrict__ WqS,
    const __bf16* __restrict__ WkS, const __bf16* __restrict__ WvS,
    __bf16* __restrict__ qb, __bf16* __restrict__ k_sw, __bf16* __restrict__ v_sw) {
  const int wave = threadIdx.x >> 5, lane = threadIdx.x & 31;
  const int nloc = lane & 15, hi = lane >> 4;
  const int half8 = hi * 8;
  const int gr0 = blockIdx.x * 16;
  const int n = gr0 >> 10, l0 = gr0 & 1023;
  const int h = wave;
  const int nh = n * NH + h;

  /* A tile: two contiguous 8-float runs per K-step, converted to bf16 */
  v16bf a[2];
  const float* xrow = x + (size_t)(gr0 + nloc) * CCH + h * HDIM;
  for (int s = 0; s < 2; s++) {
    v8f lo = *(const v8f*)(xrow + s * 32 + hi * 16);
    v8f hs = *(const v8f*)(xrow + s * 32 + hi * 16 + 8);
    for (int j = 0; j < 8; j++) {
      a[s][j] = (__bf16)lo[j];
      a[s][j + 8] = (__bf16)hs[j];
    }
  }

  v8f accq[4], acck[4], accv[4];
  wmma_proj(a[0], a[1], WqS, lane, accq);
  wmma_proj(a[0], a[1], WkS, lane, acck);
  wmma_proj(a[0], a[1], WvS, lane, accv);

  /* q: row-major [n][h][l][d] (read back as A-matrix, contiguous) */
  for (int t = 0; t < 4; t++) {
    int d = t * 16 + nloc;
    for (int r = 0; r < 8; r++) {
      int row = l0 + r + half8;
      qb[(((size_t)nh * LSEQ) + row) * HDIM + d] = (__bf16)accq[t][r];
    }
  }
  /* k: swizzled tiles [s=d/32][ct=pos/16], per (n,h) 130 tiles */
  {
    size_t kb0 = (size_t)nh * 130 * 512;
    for (int t = 0; t < 4; t++)
      for (int r = 0; r < 8; r++) {
        size_t addr = kb0 + ((size_t)(t >> 1) * 65 + (l0 >> 4)) * 512 +
                      ((t & 1) * 16 + r + half8) * 16 + nloc;
        k_sw[addr] = (__bf16)acck[t][r];
      }
  }
  /* v: swizzled tiles [sp=pos/32][dt=d/16], per (n,h) 132 tiles;
     within-lane index runs over pos -> contiguous 8-elem vector store */
  {
    size_t vb0 = (size_t)nh * 132 * 512;
    int hb = (l0 >> 4) & 1;
    for (int t = 0; t < 4; t++) {
      v8bf pk;
      for (int r = 0; r < 8; r++) pk[r] = (__bf16)accv[t][r];
      *(v8bf*)(v_sw + vb0 + ((size_t)(l0 >> 5) * 4 + t) * 512 +
               (hb * 16 + nloc) * 16 + half8) = pk;
    }
  }
}

/* ================================================================== */
/* K2: fused attention with pre/post talking heads (two-pass softmax). */
/* Block = (n, 16-query tile); wave = head.                            */
/* ================================================================== */

__global__ __launch_bounds__(256) void attn_kernel(
    const float* __restrict__ th_pre, const float* __restrict__ th_post,
    const __bf16* __restrict__ qb, const __bf16* __restrict__ k_sw,
    const __bf16* __restrict__ v_sw, __bf16* __restrict__ ao) {
  __shared__ float lds_e[8][16][32];
  __shared__ float lds_p[8][16][32];
  __shared__ float lds_t[8][16][32];

  const int wave = threadIdx.x >> 5, lane = threadIdx.x & 31;
  const int nloc = lane & 15, hi = lane >> 4;
  const int half8 = hi * 8;
  const int n = blockIdx.x >> 6, qt = blockIdx.x & 63, l0 = qt * 16;
  const int nh = n * NH + wave;
  const float slope = exp2f(-(float)(wave + 1));      /* ALiBi */
  const float invsC = 0.044194173824159216f;          /* 1/sqrt(512) */

  float thp[8], tho[8];
  for (int hh = 0; hh < 8; hh++) {
    thp[hh] = th_pre[wave * 8 + hh];
    tho[hh] = th_post[wave * 8 + hh];
  }

  v16bf aq[2];
  const __bf16* qrow = qb + (((size_t)nh * LSEQ) + l0 + nloc) * HDIM;
  for (int s = 0; s < 2; s++) {
    v8bf lo = *(const v8bf*)(qrow + s * 32 + hi * 16);
    v8bf hb = *(const v8bf*)(qrow + s * 32 + hi * 16 + 8);
    aq[s] = combine16(lo, hb);
  }

  const __bf16* ksw = k_sw + (size_t)nh * 130 * 512;
  const __bf16* vsw = v_sw + (size_t)nh * 132 * 512;

  float m_run[8], l_run[8];
  for (int r = 0; r < 8; r++) { m_run[r] = -1e30f; l_run[r] = 0.0f; }

  const int NKT = 33;  /* 33 tiles of 32 cover 1040 KV positions */

  /* ---------- pass A: exact row max / sum of mixed energies ---------- */
  for (int kt = 0; kt < NKT; kt++) {
    int c0 = kt * 32;
    for (int ct = 0; ct < 2; ct++) {
      int col = c0 + ct * 16 + nloc;
      int ctg = kt * 2 + ct; if (ctg > 64) ctg = 64;
      v8f e = v8f_zero();
      for (int s = 0; s < 2; s++) {
        v16bf bk = *(const v16bf*)(ksw + ((size_t)s * 65 + ctg) * 512 + lane * 16);
        e = WMMA_BF16(aq[s], bk, e);
      }
      for (int r = 0; r < 8; r++) {
        int row = l0 + r + half8;
        float bias = (col < LSEQ) ? (-fabsf((float)(row - col)) * slope) : 0.0f;
        lds_e[wave][r + half8][ct * 16 + nloc] = e[r] + bias;
      }
    }
    __syncthreads();
    float em[2][8];
    for (int ct = 0; ct < 2; ct++) {
      int col = c0 + ct * 16 + nloc;
      for (int r = 0; r < 8; r++) {
        float s = 0.0f;
        for (int hh = 0; hh < 8; hh++)
          s += thp[hh] * lds_e[hh][r + half8][ct * 16 + nloc];
        s *= invsC;
        if (col >= LKV) s = -1e30f;
        em[ct][r] = s;
      }
    }
    for (int r = 0; r < 8; r++) {
      float tm = fmaxf(em[0][r], em[1][r]);
      for (int off = 1; off < 16; off <<= 1)
        tm = fmaxf(tm, __shfl_xor(tm, off, 16));
      float nm = fmaxf(m_run[r], tm);
      float ps = expf(em[0][r] - nm) + expf(em[1][r] - nm);
      for (int off = 1; off < 16; off <<= 1)
        ps += __shfl_xor(ps, off, 16);
      l_run[r] = l_run[r] * expf(m_run[r] - nm) + ps;
      m_run[r] = nm;
    }
    __syncthreads();
  }
  float inv_l[8];
  for (int r = 0; r < 8; r++) inv_l[r] = 1.0f / l_run[r];

  /* ---------- pass B: probabilities, post-mix, P @ V ---------- */
  v8f acc[4];
  for (int t = 0; t < 4; t++) acc[t] = v8f_zero();

  for (int kt = 0; kt < NKT; kt++) {
    int c0 = kt * 32;
    for (int ct = 0; ct < 2; ct++) {
      int col = c0 + ct * 16 + nloc;
      int ctg = kt * 2 + ct; if (ctg > 64) ctg = 64;
      v8f e = v8f_zero();
      for (int s = 0; s < 2; s++) {
        v16bf bk = *(const v16bf*)(ksw + ((size_t)s * 65 + ctg) * 512 + lane * 16);
        e = WMMA_BF16(aq[s], bk, e);
      }
      for (int r = 0; r < 8; r++) {
        int row = l0 + r + half8;
        float bias = (col < LSEQ) ? (-fabsf((float)(row - col)) * slope) : 0.0f;
        lds_e[wave][r + half8][ct * 16 + nloc] = e[r] + bias;
      }
    }
    __syncthreads();
    for (int ct = 0; ct < 2; ct++) {
      int col = c0 + ct * 16 + nloc;
      for (int r = 0; r < 8; r++) {
        float s = 0.0f;
        for (int hh = 0; hh < 8; hh++)
          s += thp[hh] * lds_e[hh][r + half8][ct * 16 + nloc];
        s *= invsC;
        if (col >= LKV) s = -1e30f;
        lds_p[wave][r + half8][ct * 16 + nloc] = expf(s - m_run[r]) * inv_l[r];
      }
    }
    __syncthreads();
    for (int ct = 0; ct < 2; ct++)
      for (int r = 0; r < 8; r++) {
        float s = 0.0f;
        for (int hh = 0; hh < 8; hh++)
          s += tho[hh] * lds_p[hh][r + half8][ct * 16 + nloc];
        lds_t[wave][r + half8][ct * 16 + nloc] = s;
      }
    __syncthreads();
    /* A-layout read of the 16x32 post-mixed probability tile */
    const float* pt = &lds_t[wave][nloc][0];
    v8f plo = *(const v8f*)(pt + hi * 16);
    v8f phi = *(const v8f*)(pt + hi * 16 + 8);
    v16bf ap;
    for (int j = 0; j < 8; j++) {
      ap[j] = (__bf16)plo[j];
      ap[j + 8] = (__bf16)phi[j];
    }
    for (int t = 0; t < 4; t++) {
      v16bf bv = *(const v16bf*)(vsw + ((size_t)kt * 4 + t) * 512 + lane * 16);
      acc[t] = WMMA_BF16(ap, bv, acc[t]);
    }
    __syncthreads();
  }

  for (int t = 0; t < 4; t++) {
    int d = t * 16 + nloc;
    for (int r = 0; r < 8; r++) {
      int row = l0 + r + half8;
      ao[((size_t)(n * LSEQ) + row) * CCH + wave * HDIM + d] = (__bf16)acc[t][r];
    }
  }
}

/* ================================================================== */
/* K3: out @ Wo.T + bo + residual, LayerNorm1 -> x1 (f32 + bf16)       */
/* ================================================================== */

__global__ __launch_bounds__(256) void wo_ln1_kernel(
    const __bf16* __restrict__ ao, const __bf16* __restrict__ WoS,
    const float* __restrict__ bo, const float* __restrict__ x,
    const float* __restrict__ g1, const float* __restrict__ b1,
    float* __restrict__ x1f, __bf16* __restrict__ x1b) {
  __shared__ float tile[16][520];
  const int wave = threadIdx.x >> 5, lane = threadIdx.x & 31;
  const int nloc = lane & 15, hi = lane >> 4;
  const int half8 = hi * 8;
  const int gr0 = blockIdx.x * 16;

  v8f acc[4];
  for (int t = 0; t < 4; t++) acc[t] = v8f_zero();
  const __bf16* arow = ao + (size_t)(gr0 + nloc) * CCH;
  for (int s = 0; s < 16; s++) {
    v8bf lo = *(const v8bf*)(arow + s * 32 + hi * 16);
    v8bf hb = *(const v8bf*)(arow + s * 32 + hi * 16 + 8);
    v16bf a = combine16(lo, hb);
    for (int t = 0; t < 4; t++) {
      v16bf b = *(const v16bf*)(WoS + (size_t)(s * 32 + wave * 4 + t) * 512 + lane * 16);
      acc[t] = WMMA_BF16(a, b, acc[t]);
    }
  }
  for (int t = 0; t < 4; t++) {
    int col = (wave * 4 + t) * 16 + nloc;
    for (int r = 0; r < 8; r++) {
      int row = r + half8;
      tile[row][col] = acc[t][r] + bo[col] + x[(size_t)(gr0 + row) * CCH + col];
    }
  }
  __syncthreads();
  int r = threadIdx.x >> 4, sub = threadIdx.x & 15;
  float sm = 0.0f, sq = 0.0f;
  for (int i = 0; i < 32; i++) {
    float v0 = tile[r][sub + 16 * i];
    sm += v0; sq += v0 * v0;
  }
  for (int off = 1; off < 16; off <<= 1) {
    sm += __shfl_xor(sm, off, 16);
    sq += __shfl_xor(sq, off, 16);
  }
  float mu = sm * (1.0f / 512.0f);
  float var = sq * (1.0f / 512.0f) - mu * mu;
  float rs = rsqrtf(var + 1e-5f);
  for (int i = 0; i < 32; i++) {
    int col = sub + 16 * i;
    float y = (tile[r][col] - mu) * rs * g1[col] + b1[col];
    x1f[(size_t)(gr0 + r) * CCH + col] = y;
    x1b[(size_t)(gr0 + r) * CCH + col] = (__bf16)y;
  }
}

/* ================================================================== */
/* K4: conv1 (K=1536 GEMM) + SwiGLU -> hidden bf16 [8192][2048]        */
/* ================================================================== */

__global__ __launch_bounds__(256) void ffn1_kernel(
    const __bf16* __restrict__ x1b, const __bf16* __restrict__ B1s,
    const float* __restrict__ c1b, __bf16* __restrict__ hid) {
  const int wave = threadIdx.x >> 5, lane = threadIdx.x & 31;
  const int nloc = lane & 15, hi = lane >> 4;
  const int half8 = hi * 8;
  const int rt = blockIdx.x >> 3, cg = blockIdx.x & 7;
  const int gr0 = rt * 16, n = gr0 >> 10, lbase = gr0 & 1023;
  const int qidx[4] = {2 * wave, 2 * wave + 1, 2 * wave + 16, 2 * wave + 17};
  int otile[4];
  for (int t = 0; t < 4; t++) {
    int q = qidx[t];
    otile[t] = (q < 16) ? (cg * 16 + q) : (128 + cg * 16 + (q - 16));
  }
  const v8bf z8 = v8bf_zero();

  v8f acc[4];
  for (int t = 0; t < 4; t++) acc[t] = v8f_zero();

  for (int s = 0; s < 48; s++) {
    int kk0 = s * 32 + hi * 16;
    int t0 = kk0 >> 9, c0 = kk0 & 511;
    int ls0 = lbase + nloc + t0 - 1;
    v8bf lo = (ls0 >= 0 && ls0 < LSEQ)
                  ? *(const v8bf*)(x1b + (size_t)(n * LSEQ + ls0) * CCH + c0) : z8;
    int kk1 = kk0 + 8;
    int t1 = kk1 >> 9, c1 = kk1 & 511;
    int ls1 = lbase + nloc + t1 - 1;
    v8bf hb = (ls1 >= 0 && ls1 < LSEQ)
                  ? *(const v8bf*)(x1b + (size_t)(n * LSEQ + ls1) * CCH + c1) : z8;
    v16bf a = combine16(lo, hb);
    for (int t = 0; t < 4; t++) {
      v16bf b = *(const v16bf*)(B1s + ((size_t)s * 256 + otile[t]) * 512 + lane * 16);
      acc[t] = WMMA_BF16(a, b, acc[t]);
    }
  }
  for (int i = 0; i < 2; i++) {
    int o1 = cg * 256 + qidx[i] * 16 + nloc;
    int o2 = 2048 + o1;
    for (int r = 0; r < 8; r++) {
      float a1 = acc[i][r] + c1b[o1];
      float a2 = acc[i + 2][r] + c1b[o2];
      float hv = (a1 / (1.0f + expf(-a1))) * a2;   /* silu(h1) * h2 */
      int row = r + half8;
      hid[(size_t)(gr0 + row) * HIDD + o1] = (__bf16)hv;
    }
  }
}

/* ================================================================== */
/* K5: conv2 (K=6144 GEMM) + bias + residual + LayerNorm2 -> out f32   */
/* ================================================================== */

__global__ __launch_bounds__(256) void ffn2_kernel(
    const __bf16* __restrict__ hid, const __bf16* __restrict__ B2s,
    const float* __restrict__ c2b, const float* __restrict__ x1f,
    const float* __restrict__ g2, const float* __restrict__ b2v,
    float* __restrict__ out) {
  __shared__ float tile[16][520];
  const int wave = threadIdx.x >> 5, lane = threadIdx.x & 31;
  const int nloc = lane & 15, hi = lane >> 4;
  const int half8 = hi * 8;
  const int gr0 = blockIdx.x * 16, n = gr0 >> 10, lbase = gr0 & 1023;
  const v8bf z8 = v8bf_zero();

  v8f acc[4];
  for (int t = 0; t < 4; t++) acc[t] = v8f_zero();

  for (int s = 0; s < 192; s++) {
    int kk0 = s * 32 + hi * 16;
    int t0 = kk0 >> 11, c0 = kk0 & 2047;
    int ls0 = lbase + nloc + t0 - 1;
    v8bf lo = (ls0 >= 0 && ls0 < LSEQ)
                  ? *(const v8bf*)(hid + (size_t)(n * LSEQ + ls0) * HIDD + c0) : z8;
    int kk1 = kk0 + 8;
    int t1 = kk1 >> 11, c1 = kk1 & 2047;
    int ls1 = lbase + nloc + t1 - 1;
    v8bf hb = (ls1 >= 0 && ls1 < LSEQ)
                  ? *(const v8bf*)(hid + (size_t)(n * LSEQ + ls1) * HIDD + c1) : z8;
    v16bf a = combine16(lo, hb);
    for (int t = 0; t < 4; t++) {
      v16bf b = *(const v16bf*)(B2s + ((size_t)s * 32 + wave * 4 + t) * 512 + lane * 16);
      acc[t] = WMMA_BF16(a, b, acc[t]);
    }
  }
  for (int t = 0; t < 4; t++) {
    int col = (wave * 4 + t) * 16 + nloc;
    for (int r = 0; r < 8; r++) {
      int row = r + half8;
      tile[row][col] = acc[t][r] + c2b[col] + x1f[(size_t)(gr0 + row) * CCH + col];
    }
  }
  __syncthreads();
  int r = threadIdx.x >> 4, sub = threadIdx.x & 15;
  float sm = 0.0f, sq = 0.0f;
  for (int i = 0; i < 32; i++) {
    float v0 = tile[r][sub + 16 * i];
    sm += v0; sq += v0 * v0;
  }
  for (int off = 1; off < 16; off <<= 1) {
    sm += __shfl_xor(sm, off, 16);
    sq += __shfl_xor(sq, off, 16);
  }
  float mu = sm * (1.0f / 512.0f);
  float var = sq * (1.0f / 512.0f) - mu * mu;
  float rs = rsqrtf(var + 1e-5f);
  for (int i = 0; i < 32; i++) {
    int col = sub + 16 * i;
    out[(size_t)(gr0 + r) * CCH + col] =
        (tile[r][col] - mu) * rs * g2[col] + b2v[col];
  }
}

/* ================================================================== */

extern "C" void kernel_launch(void* const* d_in, const int* in_sizes, int n_in,
                              void* d_out, int out_size, void* d_ws, size_t ws_size,
                              hipStream_t stream) {
  const float* x       = (const float*)d_in[0];
  const float* Wq      = (const float*)d_in[1];
  const float* Wk      = (const float*)d_in[2];
  const float* Wv      = (const float*)d_in[3];
  const float* Wo      = (const float*)d_in[4];
  const float* bo      = (const float*)d_in[5];
  const float* th_pre  = (const float*)d_in[6];
  const float* th_post = (const float*)d_in[7];
  const float* p_keys  = (const float*)d_in[8];
  const float* p_vals  = (const float*)d_in[9];
  const float* conv1_w = (const float*)d_in[10];
  const float* conv1_b = (const float*)d_in[11];
  const float* conv2_w = (const float*)d_in[12];
  const float* conv2_b = (const float*)d_in[13];
  const float* ln1_g   = (const float*)d_in[14];
  const float* ln1_b   = (const float*)d_in[15];
  const float* ln2_g   = (const float*)d_in[16];
  const float* ln2_b   = (const float*)d_in[17];
  float* out = (float*)d_out;

  char* ws = (char*)d_ws;
  __bf16* WqS = (__bf16*)(ws + 0);                 /*   8 KB  */
  __bf16* WkS = (__bf16*)(ws + 8192);              /*   8 KB  */
  __bf16* WvS = (__bf16*)(ws + 16384);             /*   8 KB  */
  __bf16* WoS = (__bf16*)(ws + 24576);             /* 512 KB  */
  __bf16* B1s = (__bf16*)(ws + 548864);            /* 12.6 MB */
  __bf16* B2s = (__bf16*)(ws + 13131776);          /*  6.3 MB */
  __bf16* qb  = (__bf16*)(ws + 19423232);          /*  8.4 MB */
  __bf16* ksw = (__bf16*)(ws + 27811840);          /*  8.5 MB (64*130 tiles) */
  __bf16* vsw = (__bf16*)(ws + 36331520);          /*  8.7 MB (64*132 tiles) */
  __bf16* ao  = (__bf16*)(ws + 44982272);          /*  8.4 MB */
  float*  x1f = (float*) (ws + 53370880);          /* 16.8 MB */
  __bf16* x1b = (__bf16*)(ws + 70148096);          /*  8.4 MB */
  __bf16* hid = (__bf16*)(ws + 78536704);          /* 33.6 MB */

  prep_small_kernel<<<1024, 256, 0, stream>>>(Wq, Wk, Wv, Wo, p_keys, p_vals,
                                              WqS, WkS, WvS, WoS, ksw, vsw);
  prep_b1_kernel<<<24576, 256, 0, stream>>>(conv1_w, B1s);
  prep_b2_kernel<<<12288, 256, 0, stream>>>(conv2_w, B2s);

  qkv_kernel<<<512, 256, 0, stream>>>(x, WqS, WkS, WvS, qb, ksw, vsw);
  attn_kernel<<<512, 256, 0, stream>>>(th_pre, th_post, qb, ksw, vsw, ao);
  wo_ln1_kernel<<<512, 256, 0, stream>>>(ao, WoS, bo, x, ln1_g, ln1_b, x1f, x1b);
  ffn1_kernel<<<4096, 256, 0, stream>>>(x1b, B1s, conv1_b, hid);
  ffn2_kernel<<<512, 256, 0, stream>>>(hid, B2s, conv2_b, x1f, ln2_g, ln2_b, out);
}
